// MultipleGCN_17678085390507
// MI455X (gfx1250) — matrix operand order repaired and checked by reading, not compiled, not run on hardware
//
#include <hip/hip_runtime.h>
#include <hip/hip_bf16.h>
#include <stdint.h>

// ---------------------------------------------------------------------------
// MultipleGCN on MI455X (gfx1250): algebraically folded ChebConv(K=2) x2 +
// projection.  All dense math runs through v_wmma_f32_16x16x32_bf16.
// ---------------------------------------------------------------------------

typedef __attribute__((ext_vector_type(16))) __bf16          v16bf;
typedef __attribute__((ext_vector_type(8)))  __bf16          v8bf;
typedef __attribute__((ext_vector_type(8)))  float           v8f;
typedef __attribute__((ext_vector_type(4)))  float           v4f;
typedef __attribute__((ext_vector_type(8)))  unsigned short  us8;

#define NNODES 4096
#define BNROWS 32768   // B*N
#define CH     256

__device__ __forceinline__ unsigned short f2bf(float f) {
  union { float f; unsigned u; } x; x.f = f;
  unsigned r = x.u + 0x7FFFu + ((x.u >> 16) & 1u);   // round-to-nearest-even
  return (unsigned short)(r >> 16);
}

// ------------------------- degree / dis = rsqrt(deg) -----------------------
__global__ void k_dis(const float* __restrict__ adj1,
                      const float* __restrict__ adj2,
                      float* __restrict__ dis) {
  const int row = blockIdx.x;
  const float* A = (blockIdx.y == 0) ? adj1 : adj2;
  const float* arow = A + (size_t)row * NNODES;
  float s = 0.f;
  for (int c = threadIdx.x; c < NNODES; c += 256) s += arow[c];
  __shared__ float red[256];
  red[threadIdx.x] = s;
  __syncthreads();
  for (int off = 128; off > 0; off >>= 1) {
    if (threadIdx.x < off) red[threadIdx.x] += red[threadIdx.x + off];
    __syncthreads();
  }
  if (threadIdx.x == 0) {
    float deg = red[0] - arow[row];                   // remove self loop
    dis[blockIdx.y * NNODES + row] = (deg > 0.f) ? rsqrtf(deg) : 0.f;
  }
}

// ------------------- fold weights through the projection -------------------
// We = W0a@Wp[0:256] + W0b@Wp[256:512]   [256x256]
// V  = [W1a@Wp[0:256] ; W1b@Wp[256:512]] [512x256]
// be = bca@Wp_top + bcb@Wp_bot + bp      [256]
__global__ void k_fold(const float* __restrict__ W0a, const float* __restrict__ W1a,
                       const float* __restrict__ bca, const float* __restrict__ W0b,
                       const float* __restrict__ W1b, const float* __restrict__ bcb,
                       const float* __restrict__ Wp,  const float* __restrict__ bp,
                       float* __restrict__ We, float* __restrict__ V,
                       float* __restrict__ be) {
  const int c = threadIdx.x;
  const int b = blockIdx.x;
  if (b < 256) {
    const int r = b; float acc = 0.f;
    for (int k = 0; k < 256; ++k)
      acc += W0a[r*256+k]*Wp[k*256+c] + W0b[r*256+k]*Wp[(256+k)*256+c];
    We[r*256+c] = acc;
  } else if (b < 768) {
    const int r = b - 256; float acc = 0.f;
    if (r < 256) { for (int k = 0; k < 256; ++k) acc += W1a[r*256+k]*Wp[k*256+c]; }
    else { const int r2 = r-256;
           for (int k = 0; k < 256; ++k) acc += W1b[r2*256+k]*Wp[(256+k)*256+c]; }
    V[r*256+c] = acc;
  } else {
    float acc = bp[c];
    for (int k = 0; k < 256; ++k)
      acc += bca[k]*Wp[k*256+c] + bcb[k]*Wp[(256+k)*256+c];
    be[c] = acc;
  }
}

// -------- f32 row-major [M x K] -> bf16 A-fragment layout [mt][KS][32][16] --
// ISA 7.12.2 16-bit A 16x32: lanes 0-15 hold K {0..7,16..23}, lanes 16-31
// hold K {8..15,24..31}, two bf16 per VGPR.
__global__ void k_swzA(const float* __restrict__ src, unsigned short* __restrict__ dst,
                       int M, int K, int KS_total, int ksOff) {
  const int idx = blockIdx.x * 256 + threadIdx.x;     // one thread per 8 k's
  const int chunksPerRow = K >> 3;
  const int m = idx / chunksPerRow;
  const int k = (idx % chunksPerRow) << 3;
  if (m >= M) return;
  v4f a0 = *(const v4f*)(src + (size_t)m * K + k);
  v4f a1 = *(const v4f*)(src + (size_t)m * K + k + 4);
  const int kp   = k & 31;
  const int bsel = (kp >> 3) & 1;
  const int pos0 = (kp >> 4) << 3;                    // 0 or 8
  const int lane = (m & 15) + (bsel << 4);
  const int mt   = m >> 4;
  const int ks   = ksOff + (k >> 5);
  const size_t base = ((((size_t)mt * KS_total + ks) * 32 + lane) << 4) + pos0;
  us8 o;
  o[0]=f2bf(a0[0]); o[1]=f2bf(a0[1]); o[2]=f2bf(a0[2]); o[3]=f2bf(a0[3]);
  o[4]=f2bf(a1[0]); o[5]=f2bf(a1[1]); o[6]=f2bf(a1[2]); o[7]=f2bf(a1[3]);
  *(us8*)(dst + base) = o;
}

// -------- f32 row-major [K x N] -> bf16 B-fragment layout [kt][NT][32][16] --
// 16-bit B 32x16: lanes 0-15 hold K {0..7,16..23}, lanes 16-31 hold K {8..15,24..31}.
__global__ void k_swzB(const float* __restrict__ src, unsigned short* __restrict__ dst,
                       int K, int N) {
  const int idx = blockIdx.x * 256 + threadIdx.x;     // one thread per (n, 16 k's)
  const int kchunks = K >> 4;
  const int n   = idx / kchunks;
  const int k16 = idx % kchunks;
  if (n >= N) return;
  const int NT   = N >> 4;
  const int kt   = k16 >> 1;
  const int lane = (n & 15);
  const int nt   = n >> 4;
  __align__(16) unsigned short tmp[16];
  for (int p = 0; p < 16; ++p)
    tmp[p] = f2bf(src[(size_t)(k16 * 16 + p) * N + n]);
  const size_t base = ((((size_t)kt * NT + nt) * 32 + lane) << 4) + ((k16 & 1) << 3);
  *(us8*)(dst + base)       = *(const us8*)tmp;
  *(us8*)(dst + base + 256) = *(const us8*)(tmp + 8);
}

// ---- prop[z] = -(D^-1/2 (A - diag) D^-1/2) @ x0   (M=4096,K=4096,N=256) ----
// Block tile 64x256, 8 waves (2x4), wave tile 32x64; A staged via LDS bf16.
__global__ void __launch_bounds__(256) k_cheb_prop(
    const float* __restrict__ adj1, const float* __restrict__ adj2,
    const float* __restrict__ dis,  const unsigned short* __restrict__ xB,
    float* __restrict__ prop) {
  const int z = blockIdx.y;
  const float* A  = z ? adj2 : adj1;
  const float* dz = dis + z * NNODES;
  float* P = prop + (size_t)z * NNODES * CH;

  const int tid  = threadIdx.x;
  const int lane = tid & 31, wid = tid >> 5;
  const int mw = wid >> 2, nw = wid & 3;
  const int i0 = blockIdx.x * 64;

  __shared__ unsigned short ldsA[64 * 32];            // 64 rows x 32 k, bf16

  const int   sr    = tid >> 2;                       // staging row 0..63
  const int   sc    = (tid & 3) << 3;                 // staging col 0/8/16/24
  const float dis_i = dz[i0 + sr];
  const float* arow = A + (size_t)(i0 + sr) * NNODES;

  v8f acc[2][4];
  #pragma unroll
  for (int i = 0; i < 2; ++i)
    #pragma unroll
    for (int j = 0; j < 4; ++j) { v8f zz = {}; acc[i][j] = zz; }

  for (int ks = 0; ks < NNODES / 32; ++ks) {
    const int k0 = ks << 5;
    // ---- stage scaled A tile (f32 -> bf16) ----
    v4f a0 = *(const v4f*)(arow + k0 + sc);
    v4f a1 = *(const v4f*)(arow + k0 + sc + 4);
    v4f d0 = *(const v4f*)(dz + k0 + sc);
    v4f d1 = *(const v4f*)(dz + k0 + sc + 4);
    us8 pk;
    #pragma unroll
    for (int c = 0; c < 4; ++c) {
      float v0 = -dis_i * d0[c] * a0[c];
      float v1 = -dis_i * d1[c] * a1[c];
      if (i0 + sr == k0 + sc + c)     v0 = 0.f;       // diagonal removed
      if (i0 + sr == k0 + sc + 4 + c) v1 = 0.f;
      pk[c]     = f2bf(v0);
      pk[c + 4] = f2bf(v1);
    }
    __syncthreads();                                  // prior tile fully read
    *(us8*)(&ldsA[sr * 32 + sc]) = pk;
    __syncthreads();
    if (ks + 1 < NNODES / 32)
      __builtin_prefetch(arow + k0 + 32 + sc, 0, 1);  // global_prefetch next tile

    // ---- A fragments from LDS ----
    v16bf af[2];
    #pragma unroll
    for (int i = 0; i < 2; ++i) {
      const int row  = mw * 32 + i * 16 + (lane & 15);
      const int bsel = lane >> 4;
      const int base = row * 32 + 8 * bsel;           // ushort units
      union { v16bf v; v8bf h[2]; } u;
      u.h[0] = *(const v8bf*)(&ldsA[base]);
      u.h[1] = *(const v8bf*)(&ldsA[base + 16]);
      af[i] = u.v;
    }
    // ---- B fragments from pre-swizzled x0 ----
    v16bf bfr[4];
    #pragma unroll
    for (int j = 0; j < 4; ++j) {
      const int nt = nw * 4 + j;
      const size_t off = (((size_t)ks * 16 + nt) * 32 + lane) << 4;
      bfr[j] = *(const v16bf*)(xB + off);
    }
    #pragma unroll
    for (int i = 0; i < 2; ++i)
      #pragma unroll
      for (int j = 0; j < 4; ++j)
        acc[i][j] = __builtin_amdgcn_wmma_f32_16x16x32_bf16(
            false, af[i], false, bfr[j], (short)0, acc[i][j], false, false);
  }

  // ---- epilogue: C/D layout -> row-major f32 ----
  const int hi = lane >> 4, ln = lane & 15;
  #pragma unroll
  for (int i = 0; i < 2; ++i)
    #pragma unroll
    for (int j = 0; j < 4; ++j)
      #pragma unroll
      for (int v = 0; v < 8; ++v) {
        const int m = i0 + mw * 32 + i * 16 + v + 8 * hi;
        const int n = nw * 64 + j * 16 + ln;
        P[(size_t)m * CH + n] = acc[i][j][v];
      }
}

// -------- generic WMMA GEMM on pre-swizzled operands (no LDS needed) -------
// Block tile 128x128, 8 waves (4x2), wave tile 32x64.
__global__ void __launch_bounds__(256) k_gemm_swz(
    const unsigned short* __restrict__ Aswz, const unsigned short* __restrict__ Bswz,
    float* __restrict__ out, const float* __restrict__ bias,
    int N, int KS, int accumulate) {
  const int lane = threadIdx.x & 31;
  const int wid  = threadIdx.x >> 5;
  const int mw = wid >> 1, nw = wid & 1;
  const int NT = N >> 4;

  v8f acc[2][4];
  #pragma unroll
  for (int i = 0; i < 2; ++i)
    #pragma unroll
    for (int j = 0; j < 4; ++j) { v8f zz = {}; acc[i][j] = zz; }

  const int mt0 = blockIdx.x * 8 + mw * 2;
  const int nt0 = blockIdx.y * 8 + nw * 4;

  for (int ks = 0; ks < KS; ++ks) {
    v16bf a[2], b[4];
    #pragma unroll
    for (int i = 0; i < 2; ++i) {
      const size_t off = ((((size_t)(mt0 + i)) * KS + ks) * 32 + lane) << 4;
      a[i] = *(const v16bf*)(Aswz + off);
    }
    #pragma unroll
    for (int j = 0; j < 4; ++j) {
      const size_t off = (((size_t)ks * NT + (nt0 + j)) * 32 + lane) << 4;
      b[j] = *(const v16bf*)(Bswz + off);
    }
    #pragma unroll
    for (int i = 0; i < 2; ++i)
      #pragma unroll
      for (int j = 0; j < 4; ++j)
        acc[i][j] = __builtin_amdgcn_wmma_f32_16x16x32_bf16(
            false, a[i], false, b[j], (short)0, acc[i][j], false, false);
  }

  const int hi = lane >> 4, ln = lane & 15;
  const int mbase = blockIdx.x * 128 + mw * 32;
  const int nbase = blockIdx.y * 128 + nw * 64;
  #pragma unroll
  for (int i = 0; i < 2; ++i)
    #pragma unroll
    for (int j = 0; j < 4; ++j)
      #pragma unroll
      for (int v = 0; v < 8; ++v) {
        const int m = mbase + i * 16 + v + 8 * hi;
        const int n = nbase + j * 16 + ln;
        const size_t o = (size_t)m * N + n;
        float val = acc[i][j][v];
        if (bias) val += bias[n];
        if (accumulate) out[o] += val; else out[o] = val;
      }
}

// ---------------------------------------------------------------------------
extern "C" void kernel_launch(void* const* d_in, const int* in_sizes, int n_in,
                              void* d_out, int out_size, void* d_ws, size_t ws_size,
                              hipStream_t stream) {
  (void)in_sizes; (void)n_in; (void)out_size; (void)ws_size;
  const float* x    = (const float*)d_in[0];
  const float* adj1 = (const float*)d_in[1];
  const float* adj2 = (const float*)d_in[2];
  const float* W0a  = (const float*)d_in[3];
  const float* W1a  = (const float*)d_in[4];
  const float* bca  = (const float*)d_in[5];
  const float* W0b  = (const float*)d_in[6];
  const float* W1b  = (const float*)d_in[7];
  const float* bcb  = (const float*)d_in[8];
  const float* Wp   = (const float*)d_in[9];
  const float* bp   = (const float*)d_in[10];
  float* out = (float*)d_out;

  // workspace carve-up (~31 MB)
  char* ws = (char*)d_ws;
  float* dis   = (float*)ws;            ws += (size_t)2 * NNODES * 4;
  float* Wef   = (float*)ws;            ws += (size_t)256 * 256 * 4;
  float* Vf    = (float*)ws;            ws += (size_t)512 * 256 * 4;
  float* bef   = (float*)ws;            ws += 1024;
  float* prop  = (float*)ws;            ws += (size_t)2 * NNODES * CH * 4;
  unsigned short* xA    = (unsigned short*)ws; ws += (size_t)BNROWS * CH * 2;
  unsigned short* xB    = (unsigned short*)ws; ws += (size_t)NNODES * CH * 2;
  unsigned short* WeS   = (unsigned short*)ws; ws += (size_t)256 * 256 * 2;
  unsigned short* VS    = (unsigned short*)ws; ws += (size_t)512 * 256 * 2;
  unsigned short* propS = (unsigned short*)ws; ws += (size_t)NNODES * 512 * 2;

  k_dis <<<dim3(NNODES, 2), 256, 0, stream>>>(adj1, adj2, dis);
  k_fold<<<769, 256, 0, stream>>>(W0a, W1a, bca, W0b, W1b, bcb, Wp, bp, Wef, Vf, bef);

  // operand swizzles (f32 -> bf16 fragment layouts)
  k_swzA<<<(BNROWS * (CH / 8)) / 256, 256, 0, stream>>>(x, xA, BNROWS, CH, 8, 0);
  k_swzB<<<(CH * (NNODES / 16)) / 256, 256, 0, stream>>>(x, xB, NNODES, CH);
  k_swzB<<<(CH * (256 / 16)) / 256, 256, 0, stream>>>(Wef, WeS, 256, CH);
  k_swzB<<<(CH * (512 / 16)) / 256, 256, 0, stream>>>(Vf, VS, 512, CH);

  // the big normalized-adjacency GEMMs (both adjacencies, grid.y = 2)
  k_cheb_prop<<<dim3(NNODES / 64, 2), 256, 0, stream>>>(adj1, adj2, dis, xB, prop);

  // prop -> A-fragment layout, concatenated along K (K=512 total)
  k_swzA<<<(NNODES * (CH / 8)) / 256, 256, 0, stream>>>(prop, propS, NNODES, CH, 16, 0);
  k_swzA<<<(NNODES * (CH / 8)) / 256, 256, 0, stream>>>(prop + (size_t)NNODES * CH,
                                                        propS, NNODES, CH, 16, 8);

  // out = x @ We + be               (M=32768, K=256)
  k_gemm_swz<<<dim3(BNROWS / 128, 2), 256, 0, stream>>>(xA, WeS, out, bef, CH, 8, 0);
  // out[:4096] += [propA|propB] @ [Va;Vb]   (M=4096, K=512)
  k_gemm_swz<<<dim3(NNODES / 128, 2), 256, 0, stream>>>(propS, VS, out, nullptr, CH, 16, 1);
}
